// MultiScaleAttention_69020124447457
// MI455X (gfx1250) — compile-verified, matrix operand order checked by reading, not run on hardware
//
#include <hip/hip_runtime.h>
#include <hip/hip_bf16.h>

typedef __attribute__((ext_vector_type(16))) __bf16 v16bf;
typedef __attribute__((ext_vector_type(8)))  float  v8f;

#define S_LEN 2048
#define BATCH 4
#define DM 512
#define NH 8
#define DH 64
#define NTOK (S_LEN * BATCH)   // 8192

// ---------------------------------------------------------------------------
// WMMA helper: D = A(16x32 bf16) x B(32x16 bf16) + C (f32)
// ---------------------------------------------------------------------------
__device__ __forceinline__ v8f wmma_bf16(v16bf a, v16bf b, v8f c) {
  return __builtin_amdgcn_wmma_f32_16x16x32_bf16(
      /*neg_a=*/false, a, /*neg_b=*/false, b,
      /*c_mod=*/(short)0, c, /*reuse_a=*/false, /*reuse_b=*/false);
}

// A-style fragment load (16x32, 16-bit): lane holds row (lane&15),
// K = {0..7,16..23} for lanes 0-15, {8..15,24..31} for lanes 16-31.
__device__ __forceinline__ v16bf ldsFrag(const __bf16* base, int stride, int lane) {
  const __bf16* p = base + (lane & 15) * stride + ((lane >> 4) << 3);
  v16bf f;
  ((uint4*)&f)[0] = *(const uint4*)p;        // K = ko .. ko+7
  ((uint4*)&f)[1] = *(const uint4*)(p + 16); // K = ko+16 .. ko+23
  return f;
}

// ---------------------------------------------------------------------------
// CDNA5 async global->LDS copy (16B per lane), tracked by ASYNCcnt.
// Low 32 bits of a generic LDS pointer are the LDS byte offset
// (the shared aperture lives in addr[63:32], ISA 10.2).
// ---------------------------------------------------------------------------
__device__ __forceinline__ unsigned lds_u32(const void* p) {
  return (unsigned)(unsigned long long)p;
}
__device__ __forceinline__ void async_copy_b128(unsigned ldsoff, const void* g) {
  asm volatile("global_load_async_to_lds_b128 %0, %1, off"
               :: "v"(ldsoff), "v"(g) : "memory");
}
__device__ __forceinline__ void wait_async0() {
  asm volatile("s_wait_asynccnt 0x0" ::: "memory");
}

// ---------------------------------------------------------------------------
// f32 -> bf16 conversion
// ---------------------------------------------------------------------------
__global__ void cvt_bf16_kernel(const float* __restrict__ in, __bf16* __restrict__ out, int n) {
  int i = blockIdx.x * 256 + threadIdx.x;
  if (i < n) out[i] = (__bf16)in[i];
}

// ---------------------------------------------------------------------------
// Tiled WMMA GEMM: C[M x N] = A[M x K] * Bw[N x K]^T + bias[N]
//   mode 0: scatter qkv (bf16) to q/k/v  [b*H+h][S][64]
//   mode 1: outf[m*N+n]  = val   (f32)
//   mode 2: outf[m*N+n] += val   (f32)
// Block 128 threads (4 waves). Block tile 64x128, wave tile 32x64,
// K-step 32, double-buffered LDS fed by async global->LDS copies.
// ---------------------------------------------------------------------------
__global__ __launch_bounds__(128)
void gemm_bf16_kernel(const __bf16* __restrict__ A, const __bf16* __restrict__ Bw,
                      const float* __restrict__ bias, int M, int N, int K, int mode,
                      float* outf, __bf16* qo, __bf16* ko, __bf16* vo,
                      int Bdim, int H, int S) {
  __shared__ __bf16 As[2][64][40];
  __shared__ __bf16 Bs[2][128][40];

  const int tid  = threadIdx.x;
  const int lane = tid & 31;
  const int wave = tid >> 5;
  const int mBase = blockIdx.y * 64;
  const int nBase = blockIdx.x * 128;
  const int wm = (wave & 1) * 32;
  const int wn = (wave >> 1) * 64;

  v8f acc[2][4] = {};

  // staging assignment: A row = tid>>1, 16 cols; B row = tid, 32 cols
  const int arow = tid >> 1;
  const int acol = (tid & 1) * 16;

  const __bf16* agBase = A  + (size_t)(mBase + arow) * K + acol;
  const __bf16* bgBase = Bw + (size_t)(nBase + tid) * K;

  auto stage = [&](int buf, int k0) {
    unsigned la = lds_u32(&As[buf][arow][acol]);
    const __bf16* ag = agBase + k0;
    async_copy_b128(la,      ag);
    async_copy_b128(la + 16, ag + 8);
    unsigned lb = lds_u32(&Bs[buf][tid][0]);
    const __bf16* bg = bgBase + k0;
    async_copy_b128(lb,      bg);
    async_copy_b128(lb + 16, bg + 8);
    async_copy_b128(lb + 32, bg + 16);
    async_copy_b128(lb + 48, bg + 24);
  };

  stage(0, 0);
  wait_async0();
  __syncthreads();

  const int nsteps = K >> 5;
  for (int s = 0; s < nsteps; ++s) {
    const int cur = s & 1;
    const bool more = (s + 1) < nsteps;
    if (more) stage(cur ^ 1, (s + 1) << 5);   // overlap with compute below

    v16bf af[2];
#pragma unroll
    for (int mf = 0; mf < 2; ++mf)
      af[mf] = ldsFrag(&As[cur][wm + mf * 16][0], 40, lane);
#pragma unroll
    for (int nf = 0; nf < 4; ++nf) {
      v16bf bf = ldsFrag(&Bs[cur][wn + nf * 16][0], 40, lane);
#pragma unroll
      for (int mf = 0; mf < 2; ++mf)
        acc[mf][nf] = wmma_bf16(af[mf], bf, acc[mf][nf]);
    }

    if (more) wait_async0();
    __syncthreads();
  }

  const int cl = lane & 15;
  const int hs = (lane >> 4) * 8;
#pragma unroll
  for (int mf = 0; mf < 2; ++mf)
#pragma unroll
    for (int nf = 0; nf < 4; ++nf) {
      int n = nBase + wn + nf * 16 + cl;
      float bv = bias[n];
#pragma unroll
      for (int r = 0; r < 8; ++r) {
        int m = mBase + wm + mf * 16 + r + hs;
        float val = acc[mf][nf][r] + bv;
        if (mode == 0) {
          int b = m % Bdim, ssq = m / Bdim;
          int which = n >> 9, d = n & 511, h = d >> 6, di = d & 63;
          __bf16* dst = (which == 0) ? qo : ((which == 1) ? ko : vo);
          dst[(((size_t)(b * H + h) * S) + ssq) * 64 + di] = (__bf16)val;
        } else if (mode == 1) {
          outf[(size_t)m * N + n] = val;
        } else {
          outf[(size_t)m * N + n] += val;
        }
      }
    }
}

// ---------------------------------------------------------------------------
// Flash attention (bf16 WMMA, f32 online softmax).
// grid: (S/64, B*H), block 128 threads = 4 waves x 16 query rows.
// `winp`: device ptr to window size (local), or nullptr -> full S (global).
// Output written token-major: ctx[(s*B+b)*512 + h*64 + d]  (bf16).
// ---------------------------------------------------------------------------
__global__ __launch_bounds__(128)
void flash_attn_kernel(const __bf16* __restrict__ q, const __bf16* __restrict__ k,
                       const __bf16* __restrict__ v, __bf16* __restrict__ ctx,
                       int S, int Bdim, int H, const int* winp) {
  __shared__ __bf16 Ks[32][72];     // key block, row-major [key][dh]
  __shared__ __bf16 Vt[64][40];     // value block transposed [dh][key]
  __shared__ __bf16 Ps[4][16][40];  // per-wave probability tile

  const int tid = threadIdx.x, lane = tid & 31, wave = tid >> 5;
  const int bh = blockIdx.y, b = bh / H, h = bh % H;
  const int q0 = blockIdx.x * 64 + wave * 16;
  const int window = winp ? *winp : S;
  const size_t base = (size_t)bh * S * 64;
  const __bf16* qb = q + base;
  const __bf16* kb = k + base;
  const __bf16* vb = v + base;

  // Q fragments: 16 rows x dh=64 -> two A-frags, loaded straight from global.
  v16bf qf[2];
  {
    const __bf16* p = qb + (size_t)(q0 + (lane & 15)) * 64 + ((lane >> 4) << 3);
#pragma unroll
    for (int c = 0; c < 2; ++c) {
      ((uint4*)&qf[c])[0] = *(const uint4*)(p + c * 32);
      ((uint4*)&qf[c])[1] = *(const uint4*)(p + c * 32 + 16);
    }
  }

  float mrow[8], lsum[8];
#pragma unroll
  for (int r = 0; r < 8; ++r) { mrow[r] = -3.0e38f; lsum[r] = 0.f; }
  v8f o[4] = {};

  const int wstart = ((blockIdx.x * 64) / window) * window;
  const int wend = wstart + window;

  const int srow = tid >> 2;        // 0..31 (key within block)
  const int scol = (tid & 3) * 16;  // dh chunk

  for (int kb0 = wstart; kb0 < wend; kb0 += 32) {
    // --- stage K via async copy; V transposed via VGPRs --------------------
    {
      const __bf16* kp = kb + (size_t)(kb0 + srow) * 64 + scol;
      unsigned lk = lds_u32(&Ks[srow][scol]);
      async_copy_b128(lk,      kp);
      async_copy_b128(lk + 16, kp + 8);
      const __bf16* vp = vb + (size_t)(kb0 + srow) * 64 + scol;
      __bf16 tmp[16];
      *(uint4*)&tmp[0] = *(const uint4*)vp;
      *(uint4*)&tmp[8] = *(const uint4*)(vp + 8);
#pragma unroll
      for (int e = 0; e < 16; ++e) Vt[scol + e][srow] = tmp[e];
    }
    wait_async0();
    __syncthreads();

    // --- scores S = Q K^T * 1/sqrt(dh)  (16 x 32) ---------------------------
    v8f sc[2] = {};
#pragma unroll
    for (int nf = 0; nf < 2; ++nf)
#pragma unroll
      for (int c = 0; c < 2; ++c) {
        v16bf kf = ldsFrag(&Ks[nf * 16][c * 32], 72, lane);
        sc[nf] = wmma_bf16(qf[c], kf, sc[nf]);
      }
#pragma unroll
    for (int nf = 0; nf < 2; ++nf)
#pragma unroll
      for (int r = 0; r < 8; ++r) sc[nf][r] *= 0.125f;

    // --- online softmax: row stats live in one 16-lane half per row ---------
    float alpha[8];
#pragma unroll
    for (int r = 0; r < 8; ++r) {
      float t = fmaxf(sc[0][r], sc[1][r]);
#pragma unroll
      for (int off = 1; off < 16; off <<= 1) t = fmaxf(t, __shfl_xor(t, off, 32));
      float mn = fmaxf(mrow[r], t);
      alpha[r] = __expf(mrow[r] - mn);
      mrow[r] = mn;
    }
    float p0[8], p1[8];
#pragma unroll
    for (int r = 0; r < 8; ++r) {
      p0[r] = __expf(sc[0][r] - mrow[r]);
      p1[r] = __expf(sc[1][r] - mrow[r]);
      float t = p0[r] + p1[r];
#pragma unroll
      for (int off = 1; off < 16; off <<= 1) t += __shfl_xor(t, off, 32);
      lsum[r] = lsum[r] * alpha[r] + t;
    }

    // --- C-layout -> A-layout for P via per-wave LDS tile -------------------
    {
      const int cl = lane & 15, hsr = (lane >> 4) * 8;
#pragma unroll
      for (int r = 0; r < 8; ++r) {
        Ps[wave][r + hsr][cl]      = (__bf16)p0[r];
        Ps[wave][r + hsr][cl + 16] = (__bf16)p1[r];
      }
    }
    __syncthreads();

    // --- O = O*alpha + P V --------------------------------------------------
    v16bf pf = ldsFrag(&Ps[wave][0][0], 40, lane);
#pragma unroll
    for (int j = 0; j < 4; ++j) {
#pragma unroll
      for (int r = 0; r < 8; ++r) o[j][r] *= alpha[r];
      v16bf vf = ldsFrag(&Vt[j * 16][0], 40, lane);
      o[j] = wmma_bf16(pf, vf, o[j]);
    }
    __syncthreads();
  }

  // --- normalize + write token-major bf16 -----------------------------------
  const int cl = lane & 15, hsr = (lane >> 4) * 8;
#pragma unroll
  for (int r = 0; r < 8; ++r) {
    float inv = 1.f / lsum[r];
    int s = q0 + r + hsr;
    size_t t = (size_t)s * Bdim + b;
#pragma unroll
    for (int j = 0; j < 4; ++j)
      ctx[t * 512 + h * 64 + j * 16 + cl] = (__bf16)(o[j][r] * inv);
  }
}

// ---------------------------------------------------------------------------
// Host orchestration
// ---------------------------------------------------------------------------
extern "C" void kernel_launch(void* const* d_in, const int* in_sizes, int n_in,
                              void* d_out, int out_size, void* d_ws, size_t ws_size,
                              hipStream_t stream) {
  (void)in_sizes; (void)n_in; (void)out_size; (void)ws_size;

  const float* x       = (const float*)d_in[0];
  const float* w_in_g  = (const float*)d_in[1];
  const float* b_in_g  = (const float*)d_in[2];
  const float* w_out_g = (const float*)d_in[3];
  const float* b_out_g = (const float*)d_in[4];
  const float* w_in_l  = (const float*)d_in[5];
  const float* b_in_l  = (const float*)d_in[6];
  const float* w_out_l = (const float*)d_in[7];
  const float* b_out_l = (const float*)d_in[8];
  const int*   winp    = (const int*)d_in[9];
  float* out = (float*)d_out;

  // workspace layout (bf16 elements)
  __bf16* W = (__bf16*)d_ws;
  __bf16* xb    = W;                          // NTOK*DM
  __bf16* wing  = xb    + (size_t)NTOK * DM;  // 1536*512
  __bf16* winl  = wing  + (size_t)3 * DM * DM;
  __bf16* woutg = winl  + (size_t)3 * DM * DM;
  __bf16* woutl = woutg + (size_t)DM * DM;
  __bf16* qb    = woutl + (size_t)DM * DM;    // [B*H][S][64]
  __bf16* kbuf  = qb    + (size_t)BATCH * NH * S_LEN * DH;
  __bf16* vbuf  = kbuf  + (size_t)BATCH * NH * S_LEN * DH;
  __bf16* ctx   = vbuf  + (size_t)BATCH * NH * S_LEN * DH; // NTOK*DM

  auto cvt = [&](const float* src, __bf16* dst, int n) {
    cvt_bf16_kernel<<<(n + 255) / 256, 256, 0, stream>>>(src, dst, n);
  };
  cvt(x,       xb,    NTOK * DM);
  cvt(w_in_g,  wing,  3 * DM * DM);
  cvt(w_in_l,  winl,  3 * DM * DM);
  cvt(w_out_g, woutg, DM * DM);
  cvt(w_out_l, woutl, DM * DM);

  dim3 blk(128);
  dim3 gQKV(3 * DM / 128, NTOK / 64);  // (12, 128)
  dim3 gPRJ(DM / 128, NTOK / 64);      // (4, 128)
  dim3 gFA(S_LEN / 64, BATCH * NH);    // (32, 32)

  // ---- global branch ----
  gemm_bf16_kernel<<<gQKV, blk, 0, stream>>>(xb, wing, b_in_g, NTOK, 3 * DM, DM,
                                             /*mode=*/0, nullptr, qb, kbuf, vbuf,
                                             BATCH, NH, S_LEN);
  flash_attn_kernel<<<gFA, blk, 0, stream>>>(qb, kbuf, vbuf, ctx,
                                             S_LEN, BATCH, NH, /*winp=*/nullptr);
  gemm_bf16_kernel<<<gPRJ, blk, 0, stream>>>(ctx, woutg, b_out_g, NTOK, DM, DM,
                                             /*mode=*/1, out, nullptr, nullptr, nullptr,
                                             BATCH, NH, S_LEN);

  // ---- local branch (reuses q/k/v/ctx scratch) ----
  gemm_bf16_kernel<<<gQKV, blk, 0, stream>>>(xb, winl, b_in_l, NTOK, 3 * DM, DM,
                                             /*mode=*/0, nullptr, qb, kbuf, vbuf,
                                             BATCH, NH, S_LEN);
  flash_attn_kernel<<<gFA, blk, 0, stream>>>(qb, kbuf, vbuf, ctx,
                                             S_LEN, BATCH, NH, winp);
  gemm_bf16_kernel<<<gPRJ, blk, 0, stream>>>(ctx, woutl, b_out_l, NTOK, DM, DM,
                                             /*mode=*/2, out, nullptr, nullptr, nullptr,
                                             BATCH, NH, S_LEN);
}
